// BasicAttentionBlock_53068615910027
// MI455X (gfx1250) — matrix operand order repaired and checked by reading, not compiled, not run on hardware
//
#include <hip/hip_runtime.h>
#include <math.h>

typedef unsigned short ushort_t;
typedef unsigned int   uint32;

typedef __attribute__((ext_vector_type(16))) __bf16 bf16x16;
typedef __attribute__((ext_vector_type(8)))  float  f32x8;

struct __align__(16) U4 { uint32 x, y, z, w; };
union Frag16 { bf16x16 v; U4 q[2]; };

__device__ __forceinline__ ushort_t f2bf_raw(float f) {
  union { float f; uint32 u; } c; c.f = f;
  uint32 r = c.u + 0x7FFFu + ((c.u >> 16) & 1u);   // round-to-nearest-even
  return (ushort_t)(r >> 16);
}

__device__ __forceinline__ f32x8 zero8() {
  f32x8 z;
#pragma unroll
  for (int i = 0; i < 8; ++i) z[i] = 0.f;
  return z;
}

__device__ __forceinline__ f32x8 wmma_bf16(const Frag16& a, const Frag16& b, f32x8 c) {
  return __builtin_amdgcn_wmma_f32_16x16x32_bf16(false, a.v, false, b.v, (short)0, c, false, false);
}

// ---------------------------------------------------------------------------
// f32 -> bf16 conversion
// ---------------------------------------------------------------------------
__global__ void f32_to_bf16_kernel(const float* __restrict__ src,
                                   ushort_t* __restrict__ dst, int n) {
  int i = blockIdx.x * 256 + threadIdx.x;
  if (i < n) dst[i] = f2bf_raw(src[i]);
}

// ---------------------------------------------------------------------------
// LayerNorm over HID=1024, one row (b,t) per block; output bf16
// ---------------------------------------------------------------------------
__global__ __launch_bounds__(256)
void ln_kernel(const float* __restrict__ x, const float* __restrict__ g,
               const float* __restrict__ be, ushort_t* __restrict__ xn) {
  __shared__ float red[256];
  const int row = blockIdx.x;
  const int t = threadIdx.x;
  const float* xr = x + (size_t)row * 1024;
  float v[4];
  float s = 0.f;
#pragma unroll
  for (int i = 0; i < 4; ++i) { v[i] = xr[t + 256 * i]; s += v[i]; }
  red[t] = s; __syncthreads();
  for (int st = 128; st > 0; st >>= 1) { if (t < st) red[t] += red[t + st]; __syncthreads(); }
  float mean = red[0] * (1.f / 1024.f);
  __syncthreads();
  s = 0.f;
#pragma unroll
  for (int i = 0; i < 4; ++i) { float d = v[i] - mean; s += d * d; }
  red[t] = s; __syncthreads();
  for (int st = 128; st > 0; st >>= 1) { if (t < st) red[t] += red[t + st]; __syncthreads(); }
  float rinv = rsqrtf(red[0] * (1.f / 1024.f) + 1e-5f);
  ushort_t* xo = xn + (size_t)row * 1024;
#pragma unroll
  for (int i = 0; i < 4; ++i) {
    int c = t + 256 * i;
    xo[c] = f2bf_raw((v[i] - mean) * rinv * g[c] + be[c]);
  }
}

// ---------------------------------------------------------------------------
// Generic bf16 WMMA GEMM: C[M,N] = A[M,K] * B[K,N] (+bias). Row-major.
// Block tile 128x64, 8 waves, each wave 32x32. Software-pipelined: next
// k-tile's global loads issue before the current tile's WMMAs.
// ---------------------------------------------------------------------------
template <bool OUT_BF16, bool BIAS>
__global__ __launch_bounds__(256)
void gemm_bf16_kernel(const ushort_t* __restrict__ A, const ushort_t* __restrict__ B,
                      const float* __restrict__ bias, void* __restrict__ Cout,
                      int M, int N, int K) {
  __shared__ ushort_t sA[128 * 32];   // A tile, row-major [128][32]
  __shared__ ushort_t sBt[64 * 32];   // B tile transposed [n][k]
  const int t   = threadIdx.x;
  const int m0  = blockIdx.x * 128;
  const int n0  = blockIdx.y * 64;
  const int wid = t >> 5, ln = t & 31;
  const int wm  = (wid >> 1) * 32;
  const int wn  = (wid & 1) * 32;
  const int lhal = ln & 15, lhi = ln >> 4;

  f32x8 acc[2][2];
  acc[0][0] = zero8(); acc[0][1] = zero8(); acc[1][0] = zero8(); acc[1][1] = zero8();

  U4 pa[2], pb;
  auto load_tile = [&](int kb) {
#pragma unroll
    for (int e = 0; e < 2; ++e) {
      int id = t * 2 + e, row = id >> 2, c4 = id & 3;
      pa[e] = *(const U4*)&A[(size_t)(m0 + row) * K + kb + c4 * 8];
    }
    int krow = t >> 3, c8 = t & 7;
    pb = *(const U4*)&B[(size_t)(kb + krow) * N + n0 + c8 * 8];
  };
  load_tile(0);

  for (int kb = 0; kb < K; kb += 32) {
    __syncthreads();
#pragma unroll
    for (int e = 0; e < 2; ++e) {                    // commit A tile
      int id = t * 2 + e, row = id >> 2, c4 = id & 3;
      *(U4*)&sA[row * 32 + c4 * 8] = pa[e];
    }
    {                                                // commit B tile transposed
      int krow = t >> 3, c8 = t & 7;
      const ushort_t* sv = (const ushort_t*)&pb;
#pragma unroll
      for (int e2 = 0; e2 < 8; ++e2)
        sBt[(c8 * 8 + e2) * 32 + krow] = sv[e2];
    }
    __syncthreads();
    if (kb + 32 < K) load_tile(kb + 32);             // overlap next global loads

    Frag16 af[2], bfr[2];
#pragma unroll
    for (int i = 0; i < 2; ++i) {                    // A frags (16x32 lane layout)
      int m = wm + i * 16 + lhal;
      const ushort_t* p = &sA[m * 32 + lhi * 8];
      af[i].q[0] = *(const U4*)p;
      af[i].q[1] = *(const U4*)(p + 16);
    }
#pragma unroll
    for (int j = 0; j < 2; ++j) {                    // B frags (32x16 lane layout)
      int n = wn + j * 16 + lhal;
      const ushort_t* p = &sBt[n * 32 + lhi * 8];
      bfr[j].q[0] = *(const U4*)p;
      bfr[j].q[1] = *(const U4*)(p + 16);
    }
#pragma unroll
    for (int i = 0; i < 2; ++i)
#pragma unroll
      for (int j = 0; j < 2; ++j)
        acc[i][j] = wmma_bf16(af[i], bfr[j], acc[i][j]);
  }

#pragma unroll
  for (int i = 0; i < 2; ++i)
#pragma unroll
    for (int j = 0; j < 2; ++j)
#pragma unroll
      for (int r = 0; r < 8; ++r) {
        int row = m0 + wm + i * 16 + r + lhi * 8;
        int col = n0 + wn + j * 16 + lhal;
        float v = acc[i][j][r];
        if (BIAS) v += bias[col];
        if (OUT_BF16) ((ushort_t*)Cout)[(size_t)row * N + col] = f2bf_raw(v);
        else          ((float*)Cout)[(size_t)row * N + col] = v;
      }
}

// ---------------------------------------------------------------------------
// Causal flash attention, 16 heads of d=64 sharing a single 64-dim V head.
// Block = (q-superblock of 64 rows, head, batch); 4 waves x 16 q-rows.
// K staged in LDS row-major (shared by all waves, no transpose needed for
// the S-GEMM B-fragments); V staged transposed. Register-pipelined global
// loads + global_prefetch_b8 for the kb+2 tiles.
// ---------------------------------------------------------------------------
__global__ __launch_bounds__(128)
void attn_kernel(const ushort_t* __restrict__ Qg, const ushort_t* __restrict__ Kg,
                 const ushort_t* __restrict__ Vg, float* __restrict__ Oh) {
  __shared__ ushort_t sK[64 * 64];         // K tile row-major [key][d]
  __shared__ ushort_t sVt[64 * 64];        // V tile transposed [d][key]
  __shared__ ushort_t sP[4 * 16 * 64];     // per-wave P tiles [row][key]
  const int t = threadIdx.x;
  const int wid = t >> 5, ln = t & 31;
  const int lhal = ln & 15, lhi = ln >> 4;
  const int qx = blockIdx.x;               // 64-row q superblock
  const int h  = blockIdx.y;
  const int b  = blockIdx.z;
  const int qRowBase = qx * 64 + wid * 16;

  // staging geometry: K tile — each thread owns half a key row (64B = 4xU4)
  const int sKey = t >> 1;
  const int sCol = (t & 1) * 4;

  Frag16 aq[2];                            // Q A-frags (d 0..31, 32..63)
  {
    const ushort_t* qrow = Qg + ((size_t)(b * 2048 + qRowBase + lhal)) * 1024 + h * 64;
#pragma unroll
    for (int c = 0; c < 2; ++c) {
      const ushort_t* p = qrow + c * 32 + lhi * 8;
      aq[c].q[0] = *(const U4*)p;
      aq[c].q[1] = *(const U4*)(p + 16);
    }
  }

  float mst[8], lst[8];
#pragma unroll
  for (int r = 0; r < 8; ++r) { mst[r] = -INFINITY; lst[r] = 0.f; }
  f32x8 o[4];
#pragma unroll
  for (int n = 0; n < 4; ++n) o[n] = zero8();

  ushort_t* sPw = &sP[wid * 16 * 64];
  const float sc = 0.125f;                 // 1/sqrt(64)

  U4 pk[4], pv[4];
  auto load_tiles = [&](int kb) {
    const ushort_t* kg = Kg + ((size_t)(b * 2048 + kb * 64 + sKey)) * 1024 + h * 64;
#pragma unroll
    for (int e = 0; e < 4; ++e) pk[e] = *(const U4*)(kg + (sCol + e) * 8);
#pragma unroll
    for (int e = 0; e < 4; ++e) {
      int id = t * 4 + e, key = id >> 3, c8 = id & 7;
      pv[e] = *(const U4*)&Vg[(size_t)(b * 2048 + kb * 64 + key) * 64 + c8 * 8];
    }
  };
  load_tiles(0);

  for (int kb = 0; kb <= qx; ++kb) {
    __syncthreads();
#pragma unroll
    for (int e = 0; e < 4; ++e)                          // commit K (no transpose)
      *(U4*)&sK[sKey * 64 + (sCol + e) * 8] = pk[e];
#pragma unroll
    for (int e = 0; e < 4; ++e) {                        // commit V transposed
      int id = t * 4 + e, key = id >> 3, c8 = id & 7;
      const ushort_t* sv = (const ushort_t*)&pv[e];
#pragma unroll
      for (int e2 = 0; e2 < 8; ++e2)
        sVt[(c8 * 8 + e2) * 64 + key] = sv[e2];
    }
    __syncthreads();
    if (kb < qx) load_tiles(kb + 1);                     // overlap next globals
    if (kb + 2 <= qx) {                                  // prefetch kb+2
      __builtin_prefetch(
          Kg + ((size_t)(b * 2048 + (kb + 2) * 64 + sKey)) * 1024 + h * 64 + sCol * 8, 0, 3);
      if (t < 64)
        __builtin_prefetch(Vg + (size_t)(b * 2048 + (kb + 2) * 64 + t) * 64, 0, 3);
    }

    // ---- S = Q K^T (16q x 64k), two 16-key tiles at a time -----------------
    f32x8 s[4];
#pragma unroll
    for (int half = 0; half < 2; ++half) {
      Frag16 bk[2][2];
#pragma unroll
      for (int j = 0; j < 2; ++j)
#pragma unroll
        for (int c = 0; c < 2; ++c) {
          int kt = half * 2 + j;
          const ushort_t* p = &sK[(kt * 16 + lhal) * 64 + c * 32 + lhi * 8];
          bk[j][c].q[0] = *(const U4*)p;
          bk[j][c].q[1] = *(const U4*)(p + 16);
        }
#pragma unroll
      for (int j = 0; j < 2; ++j) {
        int kt = half * 2 + j;
        s[kt] = zero8();
        s[kt] = wmma_bf16(aq[0], bk[j][0], s[kt]);
        s[kt] = wmma_bf16(aq[1], bk[j][1], s[kt]);
      }
    }

    const bool diag = (kb == qx);
#pragma unroll
    for (int kt = 0; kt < 4; ++kt)
#pragma unroll
      for (int r = 0; r < 8; ++r) {
        float v = s[kt][r] * sc;
        if (diag) {
          int qrow = qRowBase + r + lhi * 8;
          int key  = kb * 64 + kt * 16 + lhal;
          if (key > qrow) v = -INFINITY;
        }
        s[kt][r] = v;
      }

#pragma unroll
    for (int r = 0; r < 8; ++r) {          // online softmax per row
      float mx = fmaxf(fmaxf(s[0][r], s[1][r]), fmaxf(s[2][r], s[3][r]));
#pragma unroll
      for (int off = 1; off < 16; off <<= 1)
        mx = fmaxf(mx, __shfl_xor(mx, off, 32));
      float mnew  = fmaxf(mst[r], mx);
      float alpha = __expf(mst[r] - mnew);
      lst[r] *= alpha;
#pragma unroll
      for (int n = 0; n < 4; ++n) o[n][r] *= alpha;
      float rs = 0.f;
#pragma unroll
      for (int kt = 0; kt < 4; ++kt) {
        float p = __expf(s[kt][r] - mnew);
        s[kt][r] = p;
        rs += p;
      }
#pragma unroll
      for (int off = 1; off < 16; off <<= 1)
        rs += __shfl_xor(rs, off, 32);
      lst[r] += rs;
      mst[r] = mnew;
    }

#pragma unroll
    for (int kt = 0; kt < 4; ++kt)         // P -> LDS (bf16), per-wave region
#pragma unroll
      for (int r = 0; r < 8; ++r) {
        int rid = r + lhi * 8;
        sPw[rid * 64 + kt * 16 + lhal] = f2bf_raw(s[kt][r]);
      }

    Frag16 ap[2];                          // P back as A-frags
#pragma unroll
    for (int c = 0; c < 2; ++c) {
      const ushort_t* p = &sPw[lhal * 64 + c * 32 + lhi * 8];
      ap[c].q[0] = *(const U4*)p;
      ap[c].q[1] = *(const U4*)(p + 16);
    }

    // ---- O += P V, two 16-dim output tiles at a time ----------------------
#pragma unroll
    for (int half = 0; half < 2; ++half) {
      Frag16 bv[2][2];
#pragma unroll
      for (int j = 0; j < 2; ++j)
#pragma unroll
        for (int c = 0; c < 2; ++c) {
          int n = half * 2 + j;
          int d = n * 16 + lhal;
          const ushort_t* p = &sVt[d * 64 + c * 32 + lhi * 8];
          bv[j][c].q[0] = *(const U4*)p;
          bv[j][c].q[1] = *(const U4*)(p + 16);
        }
#pragma unroll
      for (int j = 0; j < 2; ++j) {
        int n = half * 2 + j;
        o[n] = wmma_bf16(ap[0], bv[j][0], o[n]);
        o[n] = wmma_bf16(ap[1], bv[j][1], o[n]);
      }
    }
  }

#pragma unroll
  for (int n = 0; n < 4; ++n)
#pragma unroll
    for (int r = 0; r < 8; ++r) {
      int rid = r + lhi * 8;
      int q   = qRowBase + rid;
      int d   = n * 16 + lhal;
      Oh[(((size_t)b * 16 + h) * 2048 + q) * 64 + d] = o[n][r] / lst[r];
    }
}

// ---------------------------------------------------------------------------
// Head-mean pooling: mbar[b,t,d] = (1/16) sum_h Oh[b,h,t,d]  (bf16 out)
// ---------------------------------------------------------------------------
__global__ void headmean_kernel(const float* __restrict__ Oh, ushort_t* __restrict__ mbar) {
  int i = blockIdx.x * 256 + threadIdx.x;  // 0 .. 4096*64
  if (i >= 4096 * 64) return;
  int d = i & 63, bt = i >> 6;
  int b = bt >> 11, tt = bt & 2047;
  float s = 0.f;
  for (int h = 0; h < 16; ++h)
    s += Oh[(((size_t)b * 16 + h) * 2048 + tt) * 64 + d];
  mbar[i] = f2bf_raw(s * (1.f / 16.f));
}

// ---------------------------------------------------------------------------
// Static path (layer-2 algebraic collapse):
// add2[b] = (relu(sf[b] @ Wse + bse) @ Wv2 + bv2) @ Wo2     (all f32, tiny)
// ---------------------------------------------------------------------------
__global__ __launch_bounds__(256)
void static_kernel(const float* __restrict__ sf, const float* __restrict__ Wse,
                   const float* __restrict__ bse, const float* __restrict__ Wv2,
                   const float* __restrict__ bv2, const float* __restrict__ Wo2,
                   float* __restrict__ add2) {
  __shared__ float s_sf[1024];
  __shared__ float s_ctx[1024];
  __shared__ float s_v2[64];
  const int b = blockIdx.x, t = threadIdx.x;
  for (int i = 0; i < 4; ++i) s_sf[t + 256 * i] = sf[b * 1024 + t + 256 * i];
  __syncthreads();
  for (int i = 0; i < 4; ++i) {
    int j = t + 256 * i;
    float a = bse[j];
    for (int k = 0; k < 1024; ++k) a += s_sf[k] * Wse[(size_t)k * 1024 + j];
    s_ctx[j] = a > 0.f ? a : 0.f;
  }
  __syncthreads();
  if (t < 64) {
    float a = bv2[t];
    for (int k = 0; k < 1024; ++k) a += s_ctx[k] * Wv2[(size_t)k * 64 + t];
    s_v2[t] = a;
  }
  __syncthreads();
  for (int i = 0; i < 4; ++i) {
    int j = t + 256 * i;
    float a = 0.f;
    for (int k = 0; k < 64; ++k) a += s_v2[k] * Wo2[(size_t)k * 1024 + j];
    add2[b * 1024 + j] = a;
  }
}

// ---------------------------------------------------------------------------
// out = x + attn1_out + add2[b] (broadcast over time)
// ---------------------------------------------------------------------------
__global__ void final_kernel(const float* __restrict__ x, const float* __restrict__ attnOut,
                             const float* __restrict__ add2, float* __restrict__ out) {
  size_t i = (size_t)blockIdx.x * 256 + threadIdx.x;
  if (i >= (size_t)4096 * 1024) return;
  int c = (int)(i & 1023);
  int b = (int)(i >> 21);                  // / (2048*1024)
  out[i] = x[i] + attnOut[i] + add2[b * 1024 + c];
}

// ---------------------------------------------------------------------------
// Host launcher
// ---------------------------------------------------------------------------
extern "C" void kernel_launch(void* const* d_in, const int* in_sizes, int n_in,
                              void* d_out, int out_size, void* d_ws, size_t ws_size,
                              hipStream_t stream) {
  (void)in_sizes; (void)n_in; (void)out_size;
  const float* x   = (const float*)d_in[0];
  const float* sf  = (const float*)d_in[1];
  const float* Wq1 = (const float*)d_in[2];
  const float* bq1 = (const float*)d_in[3];
  const float* Wk1 = (const float*)d_in[4];
  const float* bk1 = (const float*)d_in[5];
  const float* Wv1 = (const float*)d_in[6];
  const float* bv1 = (const float*)d_in[7];
  const float* Wo1 = (const float*)d_in[8];
  const float* Wv2 = (const float*)d_in[13];
  const float* bv2 = (const float*)d_in[14];
  const float* Wo2 = (const float*)d_in[15];
  const float* Wse = (const float*)d_in[16];
  const float* bse = (const float*)d_in[17];
  const float* g1  = (const float*)d_in[18];
  const float* be1 = (const float*)d_in[19];
  float* out = (float*)d_out;

  // workspace carve-up (all 256B-aligned)
  char* ws = (char*)d_ws;
  constexpr size_t OFF_XN   = 0;                      // 4096*1024 bf16
  constexpr size_t OFF_WQ   = 8388608;                // 1024*1024 bf16
  constexpr size_t OFF_WK   = 10485760;
  constexpr size_t OFF_WV   = 12582912;               // 1024*64 bf16
  constexpr size_t OFF_WO   = 12713984;               // 64*1024 bf16
  constexpr size_t OFF_Q    = 12845056;               // 4096*1024 bf16
  constexpr size_t OFF_K    = 21233664;
  constexpr size_t OFF_V    = 29622272;               // 4096*64 bf16
  constexpr size_t OFF_OH   = 30146560;               // 2*16*2048*64 f32
  constexpr size_t OFF_MBAR = 46923776;               // 4096*64 bf16
  constexpr size_t OFF_AOUT = 47448064;               // 4096*1024 f32
  constexpr size_t OFF_ADD2 = 64225280;               // 2*1024 f32
  constexpr size_t NEED     = 64233472;
  if (ws_size < NEED) return;

  ushort_t* xn_bf   = (ushort_t*)(ws + OFF_XN);
  ushort_t* Wq_bf   = (ushort_t*)(ws + OFF_WQ);
  ushort_t* Wk_bf   = (ushort_t*)(ws + OFF_WK);
  ushort_t* Wv_bf   = (ushort_t*)(ws + OFF_WV);
  ushort_t* Wo_bf   = (ushort_t*)(ws + OFF_WO);
  ushort_t* Q_bf    = (ushort_t*)(ws + OFF_Q);
  ushort_t* K_bf    = (ushort_t*)(ws + OFF_K);
  ushort_t* V_bf    = (ushort_t*)(ws + OFF_V);
  float*    Oh      = (float*)(ws + OFF_OH);
  ushort_t* mbar_bf = (ushort_t*)(ws + OFF_MBAR);
  float*    attnOut = (float*)(ws + OFF_AOUT);
  float*    add2    = (float*)(ws + OFF_ADD2);

  // weight conversions
  f32_to_bf16_kernel<<<(1024 * 1024 + 255) / 256, 256, 0, stream>>>(Wq1, Wq_bf, 1024 * 1024);
  f32_to_bf16_kernel<<<(1024 * 1024 + 255) / 256, 256, 0, stream>>>(Wk1, Wk_bf, 1024 * 1024);
  f32_to_bf16_kernel<<<(1024 * 64 + 255) / 256, 256, 0, stream>>>(Wv1, Wv_bf, 1024 * 64);
  f32_to_bf16_kernel<<<(64 * 1024 + 255) / 256, 256, 0, stream>>>(Wo1, Wo_bf, 64 * 1024);

  // x1 = LN(x)  (bf16)
  ln_kernel<<<4096, 256, 0, stream>>>(x, g1, be1, xn_bf);

  // projections: Q, K, V  (bf16 out, bias)
  gemm_bf16_kernel<true, true><<<dim3(32, 16), 256, 0, stream>>>(
      xn_bf, Wq_bf, bq1, Q_bf, 4096, 1024, 1024);
  gemm_bf16_kernel<true, true><<<dim3(32, 16), 256, 0, stream>>>(
      xn_bf, Wk_bf, bk1, K_bf, 4096, 1024, 1024);
  gemm_bf16_kernel<true, true><<<dim3(32, 1), 256, 0, stream>>>(
      xn_bf, Wv_bf, bv1, V_bf, 4096, 64, 1024);

  // causal attention with shared V head, per-head outputs
  attn_kernel<<<dim3(32, 16, 2), 128, 0, stream>>>(Q_bf, K_bf, V_bf, Oh);

  // head-mean pooling then output projection
  headmean_kernel<<<(4096 * 64 + 255) / 256, 256, 0, stream>>>(Oh, mbar_bf);
  gemm_bf16_kernel<false, false><<<dim3(32, 16), 256, 0, stream>>>(
      mbar_bf, Wo_bf, nullptr, attnOut, 4096, 1024, 64);

  // layer-2 collapse: per-batch broadcast vector
  static_kernel<<<2, 256, 0, stream>>>(sf, Wse, bse, Wv2, bv2, Wo2, add2);

  // out = x + attn1_out + add2
  final_kernel<<<(4096 * 1024 + 255) / 256, 256, 0, stream>>>(x, attnOut, add2, out);
}
